// MultiHeadDistanceLayer_11158325035424
// MI455X (gfx1250) — matrix operand order repaired and checked by reading, not compiled, not run on hardware
//
#include <hip/hip_runtime.h>
#include <hip/hip_bf16.h>
#include <stdint.h>

// ---------------------------------------------------------------------------
// MultiHeadDistanceLayer for MI455X (gfx1250, wave32, WMMA f16->f32, TDM)
//
//   k_prep : x (f32) -> x16, xpe16 (f16, pos-enc added for q/k path)
//            W* (f32 [256,512]) -> Wt* (f16 transposed [512,256])
//   k_proj : (x+pe)@Wq -> Q f16 [B,H,L,64]  (1/sqrt(64) folded in)
//            (x+pe)@Wk -> K f16 [B,H,L,64]
//            x@Wv      -> V f16 [B,H,64,L]  (transposed for PV B-fragments)
//   k_attn : flash attention; K/V chunks staged into LDS by the Tensor Data
//            Mover (tensor_load_to_lds, double-buffered, s_wait_tensorcnt),
//            shared by the 4 waves of a block (all same (b,h)).
// ---------------------------------------------------------------------------

#define NUM_B   8
#define SEQ_L   1024
#define DIM_D   256
#define NUM_H   8
#define HEAD_D  64
#define HD      512
#define MROWS   8192

typedef __attribute__((ext_vector_type(16))) _Float16     v16h;
typedef __attribute__((ext_vector_type(8)))  _Float16     v8h;
typedef __attribute__((ext_vector_type(8)))  float        v8f;
typedef __attribute__((ext_vector_type(4)))  unsigned int u32x4;
typedef __attribute__((ext_vector_type(8)))  int          i32x8;
typedef __attribute__((ext_vector_type(4)))  int          i32x4;

static __device__ __forceinline__ v16h cat16(v8h a, v8h b) {
  return __builtin_shufflevector(a, b, 0,1,2,3,4,5,6,7,8,9,10,11,12,13,14,15);
}

// ---------------------------------------------------------------------------
// TDM: issue a 2-D tensor_load_to_lds.  D# bit layout per CDNA5 ISA ch.8:
//  group0: count=1 | lds_addr | global_addr[56:0] | type=2
//  group1: data_size=1(2B) | tensor_dim0/1 | tile_dim0/1 | dim0_stride
//  groups 2/3: zero (2-D tile, tile_dim2/3/4 = 0 -> unused)
// dims/strides in element (2-byte) units.
// ---------------------------------------------------------------------------
static __device__ __forceinline__ void tdm_load_2d(unsigned lds_addr,
                                                   const _Float16* gptr,
                                                   unsigned dim0, unsigned dim1,
                                                   unsigned stride0) {
  unsigned long long ga = (unsigned long long)(uintptr_t)gptr;
  u32x4 g0;
  g0[0] = 1u;                                            // count=1, user mode
  g0[1] = lds_addr;                                      // LDS byte address
  g0[2] = (unsigned)(ga & 0xFFFFFFFFu);                  // global_addr[31:0]
  g0[3] = (unsigned)((ga >> 32) & 0x01FFFFFFu) | (2u << 30); // [56:32] | type=2
  i32x8 g1;
  g1[0] = (int)(1u << 16);                               // data_size=1 (2B)
  g1[1] = (int)(dim0 << 16);                             // tensor_dim0[15:0]
  g1[2] = (int)((dim0 >> 16) | (dim1 << 16));            // dim0 hi | tensor_dim1 lo
  g1[3] = (int)((dim1 >> 16) | (dim0 << 16));            // dim1 hi | tile_dim0
  g1[4] = (int)(dim1 & 0xFFFFu);                         // tile_dim1 (tile_dim2=0)
  g1[5] = (int)stride0;                                  // tensor_dim0_stride[31:0]
  g1[6] = 0;                                             // stride0 hi | stride1 lo
  g1[7] = 0;
  i32x4 z4 = {0, 0, 0, 0};
#if __clang_major__ >= 23
  i32x8 z8 = {0, 0, 0, 0, 0, 0, 0, 0};
  __builtin_amdgcn_tensor_load_to_lds(g0, g1, z4, z4, z8, 0);
#else
  __builtin_amdgcn_tensor_load_to_lds(g0, g1, z4, z4, 0);
#endif
}

// ---------------------------------------------------------------------------
// Prep: f16 conversion, positional encoding, weight transpose
// ---------------------------------------------------------------------------
__global__ void k_prep(const float* __restrict__ x,
                       const float* __restrict__ Wq,
                       const float* __restrict__ Wk,
                       const float* __restrict__ Wv,
                       _Float16* __restrict__ x16,
                       _Float16* __restrict__ xpe16,
                       _Float16* __restrict__ wtq,
                       _Float16* __restrict__ wtk,
                       _Float16* __restrict__ wtv) {
  const int idx = blockIdx.x * blockDim.x + threadIdx.x;
  const int NX = MROWS * DIM_D;
  if (idx < NX) {
    float v = x[idx];
    int d = idx & (DIM_D - 1);
    int l = (idx >> 8) & (SEQ_L - 1);
    float inv = __expf(-9.210340371976184f * (float)(d & ~1) * (1.0f / (float)DIM_D));
    float arg = (float)l * inv;
    float pe  = (d & 1) ? __cosf(arg) : __sinf(arg);
    x16[idx]   = (_Float16)v;
    xpe16[idx] = (_Float16)(v + pe);
  }
  if (idx < 3 * HD * DIM_D) {
    int w = idx / (HD * DIM_D);
    int r = idx - w * (HD * DIM_D);
    int n = r >> 8, k = r & (DIM_D - 1);
    const float* W  = (w == 0) ? Wq : (w == 1) ? Wk : Wv;
    _Float16*    Wt = (w == 0) ? wtq : (w == 1) ? wtk : wtv;
    Wt[r] = (_Float16)W[k * HD + n];
  }
}

// ---------------------------------------------------------------------------
// Projection GEMM: one wave computes 16(M) x 64(N); K loop step 32.
// All four B-fragments loaded into distinct regs before the WMMA group so
// the scheduler can keep loads in flight (partial loadcnt waits).
// ---------------------------------------------------------------------------
__global__ void __launch_bounds__(128)
k_proj(const _Float16* __restrict__ A,    // [8192,256]
       const _Float16* __restrict__ Wt,   // [512,256]
       const float*    __restrict__ bias, // [8]
       _Float16* __restrict__ Out,
       int transposeV, float outScale) {
  const int lane = threadIdx.x & 31;
  const int wid  = threadIdx.x >> 5;
  const int task = blockIdx.x * 4 + wid;
  const int Mbase = (task >> 3) * 16;
  const int Nbase = (task & 7) * 64;
  const int m  = lane & 15;
  const int hi = lane >> 4;

  v8f acc[4] = {v8f{}, v8f{}, v8f{}, v8f{}};
  const _Float16* arow = A + (size_t)(Mbase + m) * DIM_D;

  for (int kb = 0; kb < DIM_D; kb += 32) {
    v8h a0 = *(const v8h*)(arow + kb + 8 * hi);
    v8h a1 = *(const v8h*)(arow + kb + 16 + 8 * hi);
    v16h Af = cat16(a0, a1);
    v16h Bf[4];
#pragma unroll
    for (int j = 0; j < 4; ++j) {
      const _Float16* brow = Wt + (size_t)(Nbase + j * 16 + m) * DIM_D + kb + 8 * hi;
      Bf[j] = cat16(*(const v8h*)brow, *(const v8h*)(brow + 16));
    }
#pragma unroll
    for (int j = 0; j < 4; ++j)
      acc[j] = __builtin_amdgcn_wmma_f32_16x16x32_f16(
          false, Af, false, Bf[j], (short)0, acc[j], false, false);
  }

  const int b = Mbase >> 10;
#pragma unroll
  for (int j = 0; j < 4; ++j) {
    const int ncol = Nbase + j * 16 + m;
    const int h = ncol >> 6;
    const int d = ncol & 63;
    const float bb = bias[h];
    if (transposeV) {
      v8h pk;
#pragma unroll
      for (int r = 0; r < 8; ++r)
        pk[r] = (_Float16)((acc[j][r] + bb) * outScale);
      const int l0 = (Mbase & (SEQ_L - 1)) + 8 * hi;
      *(v8h*)(Out + ((size_t)(b * NUM_H + h) * HEAD_D + d) * SEQ_L + l0) = pk;
    } else {
#pragma unroll
      for (int r = 0; r < 8; ++r) {
        const int row = Mbase + r + 8 * hi;
        const int lr  = row & (SEQ_L - 1);
        Out[((size_t)(b * NUM_H + h) * SEQ_L + lr) * HEAD_D + d] =
            (_Float16)((acc[j][r] + bb) * outScale);
      }
    }
  }
}

// ---------------------------------------------------------------------------
// Flash attention. Block = 4 waves, all the same (b,h); wave w owns query
// tile qtile*4+w. Wave 0 drives the TDM: double-buffered K [32x64] and
// V^T [64x32] tiles in LDS, overlapped with compute via TENSORcnt.
// ---------------------------------------------------------------------------
__global__ void __launch_bounds__(128)
k_attn(const _Float16* __restrict__ Q,   // [B,H,L,64] (pre-scaled 1/8)
       const _Float16* __restrict__ K,   // [B,H,L,64]
       const _Float16* __restrict__ Vt,  // [B,H,64,L]
       float* __restrict__ out) {        // [B,L,512]
  __shared__ __align__(16) _Float16 smemK[2][32][HEAD_D]; // 8 KB
  __shared__ __align__(16) _Float16 smemV[2][HEAD_D][32]; // 8 KB
  __shared__ __align__(16) _Float16 smemP[4][16][32];     // 4 KB P staging

  const int lane = threadIdx.x & 31;
  const int wid  = threadIdx.x >> 5;
  const int task = blockIdx.x * 4 + wid;   // 4096 tasks = 64 bh * 64 qtiles
  const int bh    = task >> 6;
  const int qbase = (task & 63) * 16;
  const int m  = lane & 15;
  const int hi = lane >> 4;

  const _Float16* Qb = Q  + (size_t)bh * SEQ_L * HEAD_D;
  const _Float16* Kb = K  + (size_t)bh * SEQ_L * HEAD_D;
  const _Float16* Vb = Vt + (size_t)bh * HEAD_D * SEQ_L;

  // generic->LDS address truncation: low 32 bits of an LDS generic address
  // are the LDS byte offset (ISA aperture mapping)
  const unsigned ldsK[2] = {(unsigned)(uintptr_t)&smemK[0][0][0],
                            (unsigned)(uintptr_t)&smemK[1][0][0]};
  const unsigned ldsV[2] = {(unsigned)(uintptr_t)&smemV[0][0][0],
                            (unsigned)(uintptr_t)&smemV[1][0][0]};

  // prime the pipeline: chunk 0 -> buffer 0 (one wave drives the TDM)
  if (wid == 0) {
    tdm_load_2d(ldsK[0], Kb, HEAD_D, 32, HEAD_D);   // 32 keys x 64 d
    tdm_load_2d(ldsV[0], Vb, 32, HEAD_D, SEQ_L);    // 64 d x 32 keys
  }

  // Q A-fragments, held in registers for the whole loop
  v16h Aq[2];
#pragma unroll
  for (int jd = 0; jd < 2; ++jd) {
    const _Float16* qp = Qb + (size_t)(qbase + m) * HEAD_D + jd * 32 + 8 * hi;
    Aq[jd] = cat16(*(const v8h*)qp, *(const v8h*)(qp + 16));
  }

  v8f O[4] = {v8f{}, v8f{}, v8f{}, v8f{}};
  float mrow[8], lrow[8];
#pragma unroll
  for (int r = 0; r < 8; ++r) { mrow[r] = -1e30f; lrow[r] = 0.0f; }

  for (int it = 0; it < SEQ_L / 32; ++it) {
    const int cur = it & 1;
    if (wid == 0) __builtin_amdgcn_s_wait_tensorcnt(0);  // chunk `it` landed
    __syncthreads();                                     // publish buffers[cur]
    if (wid == 0 && it + 1 < SEQ_L / 32) {               // start chunk it+1
      const int nkb = (it + 1) * 32;
      tdm_load_2d(ldsK[cur ^ 1], Kb + (size_t)nkb * HEAD_D, HEAD_D, 32, HEAD_D);
      tdm_load_2d(ldsV[cur ^ 1], Vb + nkb, 32, HEAD_D, SEQ_L);
    }

    // S = Q K^T : two 16-key tiles, K B-fragments from LDS (ds_load_b128)
    v16h Bk[4];
#pragma unroll
    for (int t = 0; t < 2; ++t)
#pragma unroll
      for (int jd = 0; jd < 2; ++jd) {
        const _Float16* kp = &smemK[cur][t * 16 + m][jd * 32 + 8 * hi];
        Bk[t * 2 + jd] = cat16(*(const v8h*)kp, *(const v8h*)(kp + 16));
      }
    v8f S[2] = {v8f{}, v8f{}};
#pragma unroll
    for (int t = 0; t < 2; ++t)
#pragma unroll
      for (int jd = 0; jd < 2; ++jd)
        S[t] = __builtin_amdgcn_wmma_f32_16x16x32_f16(
            false, Aq[jd], false, Bk[t * 2 + jd], (short)0, S[t], false, false);

    // online softmax; C-layout rows r+8*hi live in one 16-lane half
#pragma unroll
    for (int r = 0; r < 8; ++r) {
      float mx = fmaxf(S[0][r], S[1][r]);
      mx = fmaxf(mx, __shfl_xor(mx, 1, 32));
      mx = fmaxf(mx, __shfl_xor(mx, 2, 32));
      mx = fmaxf(mx, __shfl_xor(mx, 4, 32));
      mx = fmaxf(mx, __shfl_xor(mx, 8, 32));
      const float mn = fmaxf(mrow[r], mx);
      const float sc = __expf(mrow[r] - mn);
      const float p0 = __expf(S[0][r] - mn);
      const float p1 = __expf(S[1][r] - mn);
      float ps = p0 + p1;
      ps += __shfl_xor(ps, 1, 32);
      ps += __shfl_xor(ps, 2, 32);
      ps += __shfl_xor(ps, 4, 32);
      ps += __shfl_xor(ps, 8, 32);
      lrow[r] = lrow[r] * sc + ps;
      mrow[r] = mn;
#pragma unroll
      for (int j = 0; j < 4; ++j) O[j][r] *= sc;
      smemP[wid][r + 8 * hi][m]      = (_Float16)p0;
      smemP[wid][r + 8 * hi][16 + m] = (_Float16)p1;
    }

    // same-wave DS ordering before re-reading staged P as A-fragment
    asm volatile("s_wait_dscnt 0x0" ::: "memory");
    v16h Pf = cat16(*(const v8h*)&smemP[wid][m][8 * hi],
                    *(const v8h*)&smemP[wid][m][16 + 8 * hi]);

    // O += P V, V B-fragments from LDS (rows = d, contiguous keys)
    v16h Bv[4];
#pragma unroll
    for (int j = 0; j < 4; ++j) {
      const _Float16* vp = &smemV[cur][j * 16 + m][8 * hi];
      Bv[j] = cat16(*(const v8h*)vp, *(const v8h*)(vp + 16));
    }
#pragma unroll
    for (int j = 0; j < 4; ++j)
      O[j] = __builtin_amdgcn_wmma_f32_16x16x32_f16(
          false, Pf, false, Bv[j], (short)0, O[j], false, false);

    __syncthreads();   // all waves done with buffers[cur] before overwrite
  }

  // normalize and store f32 output [B, L, 512]
  const int b = bh >> 3, h = bh & 7;
#pragma unroll
  for (int r = 0; r < 8; ++r) {
    const float inv = 1.0f / lrow[r];
    const int row = qbase + r + 8 * hi;
    float* op = out + ((size_t)(b * SEQ_L + row)) * HD + h * HEAD_D + m;
#pragma unroll
    for (int j = 0; j < 4; ++j) op[j * 16] = O[j][r] * inv;
  }
}

// ---------------------------------------------------------------------------
// Workspace layout (33 MB): x16 4MB @0 | xpe16 4MB @4M | Wt* 3x256KB @8M |
// Q 8MB @9M | K 8MB @17M | Vt 8MB @25M
// ---------------------------------------------------------------------------
extern "C" void kernel_launch(void* const* d_in, const int* in_sizes, int n_in,
                              void* d_out, int out_size, void* d_ws, size_t ws_size,
                              hipStream_t stream) {
  const float* x  = (const float*)d_in[0];
  const float* Wq = (const float*)d_in[1];
  const float* bq = (const float*)d_in[2];
  const float* Wk = (const float*)d_in[3];
  const float* bk = (const float*)d_in[4];
  const float* Wv = (const float*)d_in[5];
  const float* bv = (const float*)d_in[6];

  char* ws = (char*)d_ws;
  _Float16* x16   = (_Float16*)(ws);
  _Float16* xpe16 = (_Float16*)(ws + (4u  << 20));
  _Float16* wtq   = (_Float16*)(ws + (8u  << 20));
  _Float16* wtk   = (_Float16*)(ws + (8u  << 20) + 262144);
  _Float16* wtv   = (_Float16*)(ws + (8u  << 20) + 524288);
  _Float16* Qf    = (_Float16*)(ws + (9u  << 20));
  _Float16* Kf    = (_Float16*)(ws + (17u << 20));
  _Float16* Vf    = (_Float16*)(ws + (25u << 20));

  k_prep<<<8192, 256, 0, stream>>>(x, Wq, Wk, Wv, x16, xpe16, wtq, wtk, wtv);

  k_proj<<<1024, 128, 0, stream>>>(xpe16, wtq, bq, Qf, 0, 0.125f); // Q (scaled)
  k_proj<<<1024, 128, 0, stream>>>(xpe16, wtk, bk, Kf, 0, 1.0f);   // K
  k_proj<<<1024, 128, 0, stream>>>(x16,   wtv, bv, Vf, 1, 1.0f);   // V (transposed)

  k_attn<<<1024, 128, 0, stream>>>(Qf, Kf, Vf, (float*)d_out);
}
